// GCN_56487409877354
// MI455X (gfx1250) — matrix operand order repaired and checked by reading, not compiled, not run on hardware
//
#include <hip/hip_runtime.h>
#include <math.h>

typedef __attribute__((ext_vector_type(16))) _Float16 v16h;
typedef __attribute__((ext_vector_type(8)))  _Float16 v8h;
typedef __attribute__((ext_vector_type(4)))  _Float16 v4h;
typedef __attribute__((ext_vector_type(8)))  float    v8f;

#define WAVES_PER_BLOCK 4

// ---------------------------------------------------------------------------
// Degree / normalization
// ---------------------------------------------------------------------------
__global__ void k_init_deg(float* deg, int N) {
    int i = blockIdx.x * blockDim.x + threadIdx.x;
    if (i < N) deg[i] = 1.0f;                      // self-loop
}

__global__ void k_deg_edges(const int* __restrict__ dst, float* deg, int E) {
    int e = blockIdx.x * blockDim.x + threadIdx.x;
    if (e < E) atomicAdd(&deg[dst[e]], 1.0f);
}

__global__ void k_norm(const float* __restrict__ deg, float* d_isqrt, float* d_inv, int N) {
    int i = blockIdx.x * blockDim.x + threadIdx.x;
    if (i < N) {
        float d = deg[i];
        d_isqrt[i] = rsqrtf(d);
        d_inv[i]   = 1.0f / d;
    }
}

// ---------------------------------------------------------------------------
// fp32 -> fp16 convert (for layer-1 input)
// ---------------------------------------------------------------------------
__global__ void k_f2h(const float* __restrict__ x, _Float16* __restrict__ xh, long long n) {
    long long t = (long long)blockIdx.x * blockDim.x + threadIdx.x;
    if (t < n) xh[t] = (_Float16)x[t];
}

// ---------------------------------------------------------------------------
// Weight prep: Wt[OUT][IN] (f16) = transpose of W[IN][OUT] (f32)
// ---------------------------------------------------------------------------
template<int IN, int OUT>
__global__ void k_prep_wt(const float* __restrict__ W, _Float16* __restrict__ Wt) {
    int t = blockIdx.x * blockDim.x + threadIdx.x;
    if (t < IN * OUT) {
        int o = t / IN, i = t % IN;
        Wt[t] = (_Float16)W[(size_t)i * OUT + o];
    }
}

// ---------------------------------------------------------------------------
// Dense GEMM + fused self-loop:
//   hw [N,OUT](f32) = h[N,IN](f16) @ W[IN,OUT]        (v_wmma_f32_16x16x32_f16)
//   agg[N,OUT](f32) = hw * d_inv[row]                  (epilogue, init of agg)
// A loads: per-lane contiguous 16B K-spans -> b128; out-of-range rows are
// CLAMPED to N-1 (garbage discarded by guarded stores) so there is no scalar
// load path at all. WMMA sits outside all divergence (EXEC all-1s).
// ---------------------------------------------------------------------------
template<int IN, int OUT>
__global__ __launch_bounds__(WAVES_PER_BLOCK * 32)
void k_gemm_wmma(const _Float16* __restrict__ hin, const _Float16* __restrict__ Wt,
                 const float* __restrict__ d_inv,
                 float* __restrict__ hw, float* __restrict__ agg, int N) {
    constexpr int KS = (IN + 31) / 32;
    const int  lane = threadIdx.x & 31;
    const int  wave = threadIdx.x >> 5;
    const int  row0 = (blockIdx.x * WAVES_PER_BLOCK + wave) * 16;
    const bool hi   = (lane & 16) != 0;
    const int  nloc = lane & 15;
    const bool full = (row0 + 16) <= N;
    const v8h  vzero = {};

    // ---- A fragments: always vectorized, row clamped for tail waves ----
    {
    }
    const int mrow = min(row0 + nloc, N - 1);
    const _Float16* rp = hin + (size_t)mrow * IN;
    v16h A[KS];
#pragma unroll
    for (int ks = 0; ks < KS; ++ks) {
        v8h s0, s1;
        {   // span0: K in [ks*32 + (hi?8:0), +8)
            const int K0 = ks * 32 + (hi ? 8 : 0);
            if (ks * 32 + 16 <= IN)      s0 = *(const v8h*)(rp + K0);
            else if (ks * 32 + 8 <= IN)  { v8h t = *(const v8h*)(rp + (hi ? 0 : K0)); s0 = hi ? vzero : t; }
            else                         s0 = vzero;
        }
        {   // span1: K in [ks*32 + 16 + (hi?8:0), +8)
            const int K1 = ks * 32 + 16 + (hi ? 8 : 0);
            if (ks * 32 + 32 <= IN)      s1 = *(const v8h*)(rp + K1);
            else if (ks * 32 + 24 <= IN) { v8h t = *(const v8h*)(rp + (hi ? 0 : K1)); s1 = hi ? vzero : t; }
            else                         s1 = vzero;
        }
#pragma unroll
        for (int i = 0; i < 8; ++i) { A[ks][i] = s0[i]; A[ks][i + 8] = s1[i]; }
    }

    // ---- per-row self-loop scale (broadcast across the 16 lanes of a row) ----
    const int rbase = row0 + (hi ? 8 : 0);
    float dv[8];
#pragma unroll
    for (int r = 0; r < 8; ++r) dv[r] = d_inv[min(rbase + r, N - 1)];

#pragma unroll
    for (int ct = 0; ct < OUT / 16; ++ct) {
        const int col = ct * 16 + nloc;
        const _Float16* wpw = Wt + (size_t)col * IN;
        v8f C = {};
#pragma unroll
        for (int ks = 0; ks < KS; ++ks) {
            const int kb = ks * 32 + (hi ? 8 : 0);
            v8h b0, b1;
            if (ks * 32 + 16 <= IN)      b0 = *(const v8h*)(wpw + kb);
            else if (ks * 32 + 8 <= IN)  { v8h t = *(const v8h*)(wpw + (hi ? 0 : kb)); b0 = hi ? vzero : t; }
            else                         b0 = vzero;
            if (ks * 32 + 32 <= IN)      b1 = *(const v8h*)(wpw + kb + 16);
            else if (ks * 32 + 24 <= IN) { v8h t = *(const v8h*)(wpw + (hi ? 0 : (kb + 16))); b1 = hi ? vzero : t; }
            else                         b1 = vzero;
            v16h B;
#pragma unroll
            for (int i = 0; i < 8; ++i) { B[i] = b0[i]; B[i + 8] = b1[i]; }
            C = __builtin_amdgcn_wmma_f32_16x16x32_f16(
                    false, A[ks], false, B, (short)0, C, false, false);
        }
        if (full) {
#pragma unroll
            for (int r = 0; r < 8; ++r) {
                const size_t idx = (size_t)(rbase + r) * OUT + col;
                float c = C[r];
                hw[idx]  = c;
                agg[idx] = c * dv[r];
            }
        } else {
#pragma unroll
            for (int r = 0; r < 8; ++r) {
                int mm = rbase + r;
                if (mm < N) {
                    const size_t idx = (size_t)mm * OUT + col;
                    float c = C[r];
                    hw[idx]  = c;
                    agg[idx] = c * dv[r];
                }
            }
        }
    }
}

// ---------------------------------------------------------------------------
// Edge messages: agg[dst] += hw[src] * (d_isqrt[src]*d_isqrt[dst])
// Thread = (edge, float4 chunk): coalesced gathers, L2-clustered f32 atomics.
// ---------------------------------------------------------------------------
template<int F>
__global__ void k_edge_msg(const int* __restrict__ src, const int* __restrict__ dst,
                           const float* __restrict__ d_isqrt,
                           const float* __restrict__ hw, float* __restrict__ agg, int E) {
    constexpr int PER = F / 4;
    long long t = (long long)blockIdx.x * blockDim.x + threadIdx.x;
    int e = (int)(t / PER);
    int c = (int)(t % PER);
    if (e < E) {
        int s = src[e], d = dst[e];
        float nm = d_isqrt[s] * d_isqrt[d];
        const float4 v = *(const float4*)(hw + (size_t)s * F + c * 4);
        float* dp = agg + (size_t)d * F + c * 4;
        atomicAdd(dp + 0, v.x * nm);
        atomicAdd(dp + 1, v.y * nm);
        atomicAdd(dp + 2, v.z * nm);
        atomicAdd(dp + 3, v.w * nm);
    }
}

// ---------------------------------------------------------------------------
// Finalize layer (vectorized x4): h = act(agg + b)
// T=_Float16 feeds the next WMMA GEMM, T=float feeds pooling.
// ---------------------------------------------------------------------------
template<int F, bool TANH, typename T>
__global__ void k_finalize(const float* __restrict__ agg, const float* __restrict__ b,
                           T* __restrict__ h, int N) {
    constexpr int PER = F / 4;
    long long t = (long long)blockIdx.x * blockDim.x + threadIdx.x;
    if (t < (long long)N * PER) {
        const int c = (int)(t % PER);
        const float4 v  = *(const float4*)(agg + t * 4);
        const float4 bb = *(const float4*)(b + c * 4);
        float r0 = v.x + bb.x, r1 = v.y + bb.y, r2 = v.z + bb.z, r3 = v.w + bb.w;
        if (TANH) { r0 = tanhf(r0); r1 = tanhf(r1); r2 = tanhf(r2); r3 = tanhf(r3); }
        if constexpr (sizeof(T) == 2) {
            v4h o = { (_Float16)r0, (_Float16)r1, (_Float16)r2, (_Float16)r3 };
            *(v4h*)((_Float16*)h + t * 4) = o;
        } else {
            float4 o = { r0, r1, r2, r3 };
            *(float4*)((float*)h + t * 4) = o;
        }
    }
}

// ---------------------------------------------------------------------------
// Global mean pool: LDS-staged (batch is sorted, so each 256-node block spans
// a small contiguous graph range) -> ds_add_f32 partials, tiny global flush.
// ---------------------------------------------------------------------------
__global__ void k_pool_zero(float* pooled, float* counts, int G) {
    int i = blockIdx.x * blockDim.x + threadIdx.x;
    if (i < G * 64) pooled[i] = 0.0f;
    if (i < G)      counts[i] = 0.0f;
}

__global__ __launch_bounds__(256)
void k_pool_acc(const float* __restrict__ h, const int* __restrict__ batch,
                float* pooled, int N) {
    __shared__ float acc[64 * 64];                 // 16 KB
    const int tid  = threadIdx.x;
    const int n0   = blockIdx.x * 256;
    const int nend = min(n0 + 256, N);
    const int g0   = batch[n0];
    const int g1   = batch[nend - 1];
    const int span = g1 - g0 + 1;                  // contiguous: batch sorted
    const bool use_lds = (span <= 64);             // block-uniform
    if (use_lds)
        for (int i = tid; i < span * 64; i += 256) acc[i] = 0.0f;
    __syncthreads();
    const int f   = tid & 63;
    const int sub = tid >> 6;                      // 0..3
    for (int n = n0 + sub; n < nend; n += 4) {
        int g = batch[n];
        float v = h[(size_t)n * 64 + f];
        if (use_lds) atomicAdd(&acc[(g - g0) * 64 + f], v);
        else         atomicAdd(&pooled[(size_t)g * 64 + f], v);
    }
    __syncthreads();
    if (use_lds)
        for (int i = tid; i < span * 64; i += 256)
            atomicAdd(&pooled[(size_t)(g0 + (i >> 6)) * 64 + (i & 63)], acc[i]);
}

__global__ void k_count_acc(const int* __restrict__ batch, float* counts, int N) {
    int i = blockIdx.x * blockDim.x + threadIdx.x;
    if (i < N) atomicAdd(&counts[batch[i]], 1.0f);
}

__global__ void k_final_linear(const float* __restrict__ pooled, const float* __restrict__ counts,
                               const float* __restrict__ Wl, const float* __restrict__ bl,
                               float* __restrict__ out, int G) {
    int t = blockIdx.x * blockDim.x + threadIdx.x;
    if (t < G * 4) {
        int g = t / 4, j = t % 4;
        float inv = 1.0f / fmaxf(counts[g], 1.0f);
        float acc = bl[j];
#pragma unroll
        for (int f = 0; f < 64; ++f)
            acc += pooled[(size_t)g * 64 + f] * inv * Wl[f * 4 + j];
        out[t] = acc;
    }
}

// ---------------------------------------------------------------------------
// Orchestration
// ---------------------------------------------------------------------------
static inline int cdiv(long long a, int b) { return (int)((a + b - 1) / b); }

extern "C" void kernel_launch(void* const* d_in, const int* in_sizes, int n_in,
                              void* d_out, int out_size, void* d_ws, size_t ws_size,
                              hipStream_t stream) {
    const float* x     = (const float*)d_in[0];
    const int*   ei    = (const int*)  d_in[1];
    const int*   batch = (const int*)  d_in[2];
    const float* W1 = (const float*)d_in[3];  const float* b1 = (const float*)d_in[4];
    const float* W2 = (const float*)d_in[5];  const float* b2 = (const float*)d_in[6];
    const float* W3 = (const float*)d_in[7];  const float* b3 = (const float*)d_in[8];
    const float* W4 = (const float*)d_in[9];  const float* b4 = (const float*)d_in[10];
    const float* Wl = (const float*)d_in[11]; const float* bl = (const float*)d_in[12];
    float* out = (float*)d_out;

    const int N = in_sizes[0] / 8;
    const int E = in_sizes[1] / 2;
    const int G = out_size / 4;
    const int* src = ei;
    const int* dst = ei + E;

    // workspace layout (256B-aligned slices)
    char* wp = (char*)d_ws;
    auto alloc = [&](size_t bytes) -> void* {
        void* r = (void*)wp;
        wp += (bytes + 255) & ~(size_t)255;
        return r;
    };
    float*    deg     = (float*)   alloc((size_t)N * 4);
    float*    d_isqrt = (float*)   alloc((size_t)N * 4);
    float*    d_inv   = (float*)   alloc((size_t)N * 4);
    float*    hw      = (float*)   alloc((size_t)N * 64 * 4);
    float*    agg     = (float*)   alloc((size_t)N * 64 * 4);
    _Float16* hh      = (_Float16*)alloc((size_t)N * 64 * 2);  // f16 activations
    _Float16* wt      = (_Float16*)alloc((size_t)64 * 64 * 2); // f16 transposed weights
    float*    pooled  = (float*)   alloc((size_t)G * 64 * 4);
    float*    counts  = (float*)   alloc((size_t)G * 4);
    (void)ws_size; (void)n_in;

    const int BT = 256;
    const int gemm_blocks = cdiv((N + 15) / 16, WAVES_PER_BLOCK);
    const int GB = WAVES_PER_BLOCK * 32;

    // --- degree + norms ---
    k_init_deg <<<cdiv(N, BT), BT, 0, stream>>>(deg, N);
    k_deg_edges<<<cdiv(E, BT), BT, 0, stream>>>(dst, deg, E);
    k_norm     <<<cdiv(N, BT), BT, 0, stream>>>(deg, d_isqrt, d_inv, N);

    // --- layer-1 input to f16 ---
    k_f2h<<<cdiv((long long)N * 8, BT), BT, 0, stream>>>(x, hh, (long long)N * 8);

    // --- layer 1: 8 -> 8, tanh ---
    k_prep_wt<8, 8><<<1, BT, 0, stream>>>(W1, wt);
    k_gemm_wmma<8, 8><<<gemm_blocks, GB, 0, stream>>>(hh, wt, d_inv, hw, agg, N);
    k_edge_msg<8><<<cdiv((long long)E * 2, BT), BT, 0, stream>>>(src, dst, d_isqrt, hw, agg, E);
    k_finalize<8, true, _Float16><<<cdiv((long long)N * 2, BT), BT, 0, stream>>>(agg, b1, hh, N);

    // --- layer 2: 8 -> 16, tanh ---
    k_prep_wt<8, 16><<<1, BT, 0, stream>>>(W2, wt);
    k_gemm_wmma<8, 16><<<gemm_blocks, GB, 0, stream>>>(hh, wt, d_inv, hw, agg, N);
    k_edge_msg<16><<<cdiv((long long)E * 4, BT), BT, 0, stream>>>(src, dst, d_isqrt, hw, agg, E);
    k_finalize<16, true, _Float16><<<cdiv((long long)N * 4, BT), BT, 0, stream>>>(agg, b2, hh, N);

    // --- layer 3: 16 -> 32, tanh ---
    k_prep_wt<16, 32><<<cdiv(16 * 32, BT), BT, 0, stream>>>(W3, wt);
    k_gemm_wmma<16, 32><<<gemm_blocks, GB, 0, stream>>>(hh, wt, d_inv, hw, agg, N);
    k_edge_msg<32><<<cdiv((long long)E * 8, BT), BT, 0, stream>>>(src, dst, d_isqrt, hw, agg, E);
    k_finalize<32, true, _Float16><<<cdiv((long long)N * 8, BT), BT, 0, stream>>>(agg, b3, hh, N);

    // --- layer 4: 32 -> 64, no tanh (write f32 into hw for pooling) ---
    k_prep_wt<32, 64><<<cdiv(32 * 64, BT), BT, 0, stream>>>(W4, wt);
    k_gemm_wmma<32, 64><<<gemm_blocks, GB, 0, stream>>>(hh, wt, d_inv, hw, agg, N);
    k_edge_msg<64><<<cdiv((long long)E * 16, BT), BT, 0, stream>>>(src, dst, d_isqrt, hw, agg, E);
    k_finalize<64, false, float><<<cdiv((long long)N * 16, BT), BT, 0, stream>>>(agg, b4, hw, N);

    // --- global mean pool + final linear ---
    k_pool_zero   <<<cdiv(G * 64, BT), BT, 0, stream>>>(pooled, counts, G);
    k_pool_acc    <<<cdiv(N, 256), 256, 0, stream>>>(hw, batch, pooled, N);
    k_count_acc   <<<cdiv(N, BT), BT, 0, stream>>>(batch, counts, N);
    k_final_linear<<<cdiv(G * 4, BT), BT, 0, stream>>>(pooled, counts, Wl, bl, out, G);
}
